// SimpleSSM_52243982188791
// MI455X (gfx1250) — matrix operand order repaired and checked by reading, not compiled, not run on hardware
//
#include <hip/hip_runtime.h>
#include <hip/hip_bf16.h>

// ---------------------------------------------------------------------------
// SimpleSSM for MI455X (gfx1250): f16 WMMA GEMMs + tiny serial scan.
// in_proj (137 GFLOP) and out_proj (69 GFLOP) dominate -> v_wmma_f32_16x16x32_f16.
// GEMM operands (x:33MB, W_in:8MB, W_out:4MB in f16) are 192MB-L2 resident.
// Inner loops: 2 moving base pointers, all other addressing via 24-bit
// immediate offsets; single live B-fragment to keep VGPRs < spill threshold.
// ---------------------------------------------------------------------------

#define BATCH   4
#define T_LEN   4096
#define D_MODEL 1024
#define D_STATE 16
#define D_INNER 2048
#define M_ROWS  (BATCH * T_LEN)          // 16384
#define N_IN    (2 * D_INNER)            // 4096

typedef __attribute__((ext_vector_type(16))) _Float16 v16h;
typedef __attribute__((ext_vector_type(8)))  float    v8f;
typedef __attribute__((ext_vector_type(4)))  unsigned v4u;

union Frag { v16h v; v4u q[2]; };

__device__ __forceinline__ v8f wmma_f16(const Frag& a, const Frag& b, v8f c) {
    return __builtin_amdgcn_wmma_f32_16x16x32_f16(
        /*neg_a=*/false, a.v, /*neg_b=*/false, b.v,
        /*c_mod=*/(short)0, c, /*reuse_a=*/false, /*reuse_b=*/false);
}

// A fragment (16x32 tile): lane L<16 -> row L, K halves {0..7, 16..23};
// lane L>=16 -> row L-16, K halves {8..15, 24..31}.  (ISA 7.12.2)
// B fragment (32x16 tile, B[k][n]=W[n][k]): lane L -> col L&15,
// same K halves as A: {0..7, 16..23} (lanes<16) or {8..15, 24..31} (lanes>=16).

// ---------------------------------------------------------------------------
__global__ void k_f32_to_f16(const float* __restrict__ src,
                             _Float16* __restrict__ dst, int n) {
    int i = blockIdx.x * blockDim.x + threadIdx.x;
    if (i < n) dst[i] = (_Float16)src[i];
}

// ---------------------------------------------------------------------------
// in_proj: xz = x @ W_in^T + b_in ; SiLU both halves; split into x_proj / gate.
// Per-wave tile 32(M) x 64(N); K = D_MODEL, step 32; 8 WMMAs / iter.
__global__ __launch_bounds__(128, 1)
void k_in_proj(const _Float16* __restrict__ A,   // x f16 (M x 1024)
               const _Float16* __restrict__ W,   // W_in f16 (4096 x 1024)
               const float* __restrict__ bias,   // b_in (4096)
               _Float16* __restrict__ xp,        // x_proj (M x 2048)
               _Float16* __restrict__ sg) {      // silu(z) (M x 2048)
    const int TN = N_IN / 64;                    // 64 tiles in N
    int wid  = blockIdx.x * (blockDim.x >> 5) + (threadIdx.x >> 5);
    int lane = threadIdx.x & 31;
    int l15  = lane & 15;
    int hi   = (lane >> 4) & 1;
    int tm = wid / TN, tn = wid % TN;
    int m0 = tm * 32, n0 = tn * 64;

    // Moving base pointers; all other addressing folds into immediate offsets.
    const _Float16* pa = A + (size_t)(m0 + l15) * D_MODEL + hi * 8;
    const _Float16* pb = W + (size_t)(n0 + l15) * D_MODEL + hi * 8;

    v8f acc[2][4] = {};
#pragma unroll 2
    for (int k = 0; k < D_MODEL; k += 32) {
        Frag a0, a1;
        a0.q[0] = *reinterpret_cast<const v4u*>(pa);
        a0.q[1] = *reinterpret_cast<const v4u*>(pa + 16);
        a1.q[0] = *reinterpret_cast<const v4u*>(pa + 16 * D_MODEL);
        a1.q[1] = *reinterpret_cast<const v4u*>(pa + 16 * D_MODEL + 16);
#pragma unroll
        for (int j = 0; j < 4; ++j) {
            Frag b;
            b.q[0] = *reinterpret_cast<const v4u*>(pb + j * 16 * D_MODEL);
            b.q[1] = *reinterpret_cast<const v4u*>(pb + j * 16 * D_MODEL + 16);
            acc[0][j] = wmma_f16(a0, b, acc[0][j]);
            acc[1][j] = wmma_f16(a1, b, acc[1][j]);
        }
        pa += 32;
        pb += 32;
    }

    // C/D layout: VGPR r -> M = r (lanes 0-15) or r+8 (lanes 16-31); N = lane&15.
    int colbase = n0 + l15;
    bool lo_half = (n0 < D_INNER);               // uniform per 64-wide tile
    _Float16* dstbase = lo_half ? xp : sg;
    int c0 = colbase - (lo_half ? 0 : D_INNER);
#pragma unroll
    for (int i = 0; i < 2; ++i) {
        int mb = m0 + 16 * i + (hi ? 8 : 0);
        _Float16* pd = dstbase + (size_t)mb * D_INNER + c0;
#pragma unroll
        for (int j = 0; j < 4; ++j) {
            float bs = bias[colbase + 16 * j];
#pragma unroll
            for (int r = 0; r < 8; ++r) {
                float v = acc[i][j][r] + bs;
                float s = v / (1.0f + __expf(-v));          // SiLU
                pd[(size_t)r * D_INNER + 16 * j] = (_Float16)s;
            }
        }
    }
}

// ---------------------------------------------------------------------------
// b = x_proj @ W_B^T : M x 16, K = 2048. One 16x16 tile per wave.
__global__ __launch_bounds__(128, 1)
void k_bproj(const _Float16* __restrict__ A,   // x_proj (M x 2048)
             const _Float16* __restrict__ WB,  // W_B f16 (16 x 2048)
             float* __restrict__ bout) {       // b (M x 16) f32
    int wid  = blockIdx.x * (blockDim.x >> 5) + (threadIdx.x >> 5);
    int lane = threadIdx.x & 31;
    int l15  = lane & 15;
    int hi   = (lane >> 4) & 1;
    int m0 = wid * 16;

    const _Float16* pa = A + (size_t)(m0 + l15) * D_INNER + hi * 8;
    const _Float16* pb = WB + (size_t)l15 * D_INNER + hi * 8;

    v8f acc = {};
#pragma unroll 4
    for (int k = 0; k < D_INNER; k += 32) {
        Frag a, b;
        a.q[0] = *reinterpret_cast<const v4u*>(pa);
        a.q[1] = *reinterpret_cast<const v4u*>(pa + 16);
        b.q[0] = *reinterpret_cast<const v4u*>(pb);
        b.q[1] = *reinterpret_cast<const v4u*>(pb + 16);
        acc = wmma_f16(a, b, acc);
        pa += 32;
        pb += 32;
    }
    int mb = m0 + (hi ? 8 : 0);
    float* po = bout + (size_t)mb * D_STATE + l15;
#pragma unroll
    for (int r = 0; r < 8; ++r)
        po[(size_t)r * D_STATE] = acc[r];
}

// ---------------------------------------------------------------------------
// Serial fixed-decay scan: h = 0.9h + 0.1*b_t per (batch,state). 64 lanes total.
__global__ void k_scan(const float* __restrict__ b, float* __restrict__ hs) {
    int tid = threadIdx.x;                 // 0..63
    int bi = tid >> 4, si = tid & 15;
    const float* bp = b  + (size_t)bi * T_LEN * D_STATE + si;
    float*       hp = hs + (size_t)bi * T_LEN * D_STATE + si;
    float h = 0.0f;
    for (int t = 0; t < T_LEN; ++t) {
        h = fmaf(0.9f, h, 0.1f * bp[(size_t)t * D_STATE]);
        hp[(size_t)t * D_STATE] = h;
    }
}

// ---------------------------------------------------------------------------
// y = (hs @ W_C^T + D * x_proj) * silu(z); y overwrites xp (f16) in place.
__global__ void k_cmix(const float* __restrict__ hs,     // (M x 16)
                       const float* __restrict__ W_C,    // (2048 x 16) f32
                       const float* __restrict__ Dv,     // (2048)
                       _Float16* __restrict__ xp,        // in: x_proj, out: y
                       const _Float16* __restrict__ sg) {
    int m = blockIdx.x;
    __shared__ float sh[D_STATE];
    if (threadIdx.x < D_STATE) sh[threadIdx.x] = hs[(size_t)m * D_STATE + threadIdx.x];
    __syncthreads();
    for (int e = threadIdx.x; e < D_INNER; e += blockDim.x) {
        const float* wc = W_C + (size_t)e * D_STATE;
        float c = 0.0f;
#pragma unroll
        for (int s = 0; s < D_STATE; ++s) c = fmaf(sh[s], wc[s], c);
        size_t idx = (size_t)m * D_INNER + e;
        float y = (c + Dv[e] * (float)xp[idx]) * (float)sg[idx];
        xp[idx] = (_Float16)y;
    }
}

// ---------------------------------------------------------------------------
// out = y @ W_out^T + b_out : M x 1024, K = 2048. 32x64 per-wave tile, f32 out.
__global__ __launch_bounds__(128, 1)
void k_out_proj(const _Float16* __restrict__ A,   // y (M x 2048)
                const _Float16* __restrict__ W,   // W_out f16 (1024 x 2048)
                const float* __restrict__ bias,   // b_out (1024)
                float* __restrict__ out) {        // (M x 1024) f32
    const int TN = D_MODEL / 64;                  // 16
    int wid  = blockIdx.x * (blockDim.x >> 5) + (threadIdx.x >> 5);
    int lane = threadIdx.x & 31;
    int l15  = lane & 15;
    int hi   = (lane >> 4) & 1;
    int tm = wid / TN, tn = wid % TN;
    int m0 = tm * 32, n0 = tn * 64;

    const _Float16* pa = A + (size_t)(m0 + l15) * D_INNER + hi * 8;
    const _Float16* pb = W + (size_t)(n0 + l15) * D_INNER + hi * 8;

    v8f acc[2][4] = {};
#pragma unroll 2
    for (int k = 0; k < D_INNER; k += 32) {
        Frag a0, a1;
        a0.q[0] = *reinterpret_cast<const v4u*>(pa);
        a0.q[1] = *reinterpret_cast<const v4u*>(pa + 16);
        a1.q[0] = *reinterpret_cast<const v4u*>(pa + 16 * D_INNER);
        a1.q[1] = *reinterpret_cast<const v4u*>(pa + 16 * D_INNER + 16);
#pragma unroll
        for (int j = 0; j < 4; ++j) {
            Frag b;
            b.q[0] = *reinterpret_cast<const v4u*>(pb + j * 16 * D_INNER);
            b.q[1] = *reinterpret_cast<const v4u*>(pb + j * 16 * D_INNER + 16);
            acc[0][j] = wmma_f16(a0, b, acc[0][j]);
            acc[1][j] = wmma_f16(a1, b, acc[1][j]);
        }
        pa += 32;
        pb += 32;
    }

    int colbase = n0 + l15;
#pragma unroll
    for (int i = 0; i < 2; ++i) {
        int mb = m0 + 16 * i + (hi ? 8 : 0);
        float* po = out + (size_t)mb * D_MODEL + colbase;
#pragma unroll
        for (int j = 0; j < 4; ++j) {
            float bs = bias[colbase + 16 * j];
#pragma unroll
            for (int r = 0; r < 8; ++r)
                po[(size_t)r * D_MODEL + 16 * j] = acc[i][j][r] + bs;
        }
    }
}

// ---------------------------------------------------------------------------
extern "C" void kernel_launch(void* const* d_in, const int* in_sizes, int n_in,
                              void* d_out, int out_size, void* d_ws, size_t ws_size,
                              hipStream_t stream) {
    const float* x     = (const float*)d_in[0];
    const float* W_in  = (const float*)d_in[1];
    const float* b_in  = (const float*)d_in[2];
    const float* W_B   = (const float*)d_in[3];
    const float* W_C   = (const float*)d_in[4];
    const float* Dv    = (const float*)d_in[5];
    const float* W_out = (const float*)d_in[6];
    const float* b_out = (const float*)d_in[7];
    float* out = (float*)d_out;

    // Workspace layout (64KB-aligned offsets), ~182.5 MB total.
    char* ws = (char*)d_ws;
    const size_t XB_N  = (size_t)M_ROWS * D_MODEL;     // 16.7M
    const size_t WI_N  = (size_t)N_IN * D_MODEL;       // 4.19M
    const size_t WB_N  = (size_t)D_STATE * D_INNER;    // 32K
    const size_t WO_N  = (size_t)D_MODEL * D_INNER;    // 2.1M
    const size_t XP_N  = (size_t)M_ROWS * D_INNER;     // 33.5M
    const size_t BS_N  = (size_t)M_ROWS * D_STATE;     // 262K

    size_t off = 0;
    _Float16* xb   = (_Float16*)(ws + off); off += XB_N * 2;     //  33.6 MB
    _Float16* Wi_h = (_Float16*)(ws + off); off += WI_N * 2;     //   8.4 MB
    _Float16* WB_h = (_Float16*)(ws + off); off += WB_N * 2;     //  64 KB
    _Float16* Wo_h = (_Float16*)(ws + off); off += WO_N * 2;     //   4.2 MB
    _Float16* xp   = (_Float16*)(ws + off); off += XP_N * 2;     //  67.1 MB (x_proj -> y)
    _Float16* sg   = (_Float16*)(ws + off); off += XP_N * 2;     //  67.1 MB
    float*    bbuf = (float*)   (ws + off); off += BS_N * 4;     //   1.0 MB
    float*    hbuf = (float*)   (ws + off); off += BS_N * 4;     //   1.0 MB
    (void)ws_size; (void)in_sizes; (void)n_in; (void)out_size;

    // 1) f32 -> f16 conversions (one pass each; thereafter L2-resident)
    k_f32_to_f16<<<(unsigned)((XB_N + 255) / 256), 256, 0, stream>>>(x, xb, (int)XB_N);
    k_f32_to_f16<<<(unsigned)((WI_N + 255) / 256), 256, 0, stream>>>(W_in, Wi_h, (int)WI_N);
    k_f32_to_f16<<<(unsigned)((WB_N + 255) / 256), 256, 0, stream>>>(W_B, WB_h, (int)WB_N);
    k_f32_to_f16<<<(unsigned)((WO_N + 255) / 256), 256, 0, stream>>>(W_out, Wo_h, (int)WO_N);

    // 2) in_proj GEMM + bias + SiLU + split: 32768 wave-tiles, 4 waves/block
    k_in_proj<<<(M_ROWS / 32) * (N_IN / 64) / 4, 128, 0, stream>>>(xb, Wi_h, b_in, xp, sg);

    // 3) B-projection GEMM: 1024 wave-tiles
    k_bproj<<<(M_ROWS / 16) / 4, 128, 0, stream>>>(xp, WB_h, bbuf);

    // 4) serial scan, 64 lanes (B x D_STATE)
    k_scan<<<1, 64, 0, stream>>>(bbuf, hbuf);

    // 5) C-projection + skip + gate (y overwrites xp)
    k_cmix<<<M_ROWS, 256, 0, stream>>>(hbuf, W_C, Dv, xp, sg);

    // 6) out_proj GEMM + bias -> f32 output
    k_out_proj<<<(M_ROWS / 32) * (D_MODEL / 64) / 4, 128, 0, stream>>>(xp, Wo_h, b_out, out);
}
